// FeatureCorrelation_63333587746895
// MI455X (gfx1250) — matrix operand order repaired and checked by reading, not compiled, not run on hardware
//
#include <hip/hip_runtime.h>
#include <hip/hip_bf16.h>
#include <stdint.h>

typedef __attribute__((ext_vector_type(16))) _Float16 v16h;
typedef __attribute__((ext_vector_type(8)))  float    v8f;

#define Bn   8
#define Cn   256
#define HWn  4096
#define OUTC 256

union FragH {
  v16h  h;
  uint4 q[2];
};

// ---------------------------------------------------------------------------
// Kernel P: fold conv bias + BN into per-output-channel affine, Wsum trick,
// convert W[:, 0:256] to f16.
// ---------------------------------------------------------------------------
__global__ __launch_bounds__(256) void prep_kernel(
    const float* __restrict__ conv_w, const float* __restrict__ conv_b,
    const float* __restrict__ gamma,  const float* __restrict__ beta,
    const float* __restrict__ mean,   const float* __restrict__ var,
    _Float16* __restrict__ Wh, float* __restrict__ w_scale,
    float* __restrict__ w_wsum_s, float* __restrict__ w_const)
{
  int o = threadIdx.x;  // 256 threads, 1 block
  const float* row = conv_w + (size_t)o * 512;
  float wsum = 0.f;
  #pragma unroll 4
  for (int c = 0; c < 256; ++c) wsum += row[256 + c];
  #pragma unroll 4
  for (int c = 0; c < 256; ++c) Wh[o * 256 + c] = (_Float16)row[c];
  float sc = gamma[o] * rsqrtf(var[o] + 1e-5f);
  w_scale[o]  = sc;
  w_wsum_s[o] = wsum * sc;
  w_const[o]  = conv_b[o] * sc + beta[o] - mean[o] * sc;
}

// ---------------------------------------------------------------------------
// Kernel N1: per (b, spatial i) L2 norms over 256 channels (clamped at eps),
// plus inverses.  Adjacent threads -> adjacent i -> coalesced strided reads.
// ---------------------------------------------------------------------------
__global__ __launch_bounds__(256) void norm_kernel(
    const float* __restrict__ src, const float* __restrict__ tgt,
    float* __restrict__ norm_src, float* __restrict__ inv_src,
    float* __restrict__ inv_tgt)
{
  int tid = blockIdx.x * 256 + threadIdx.x;    // flat (b,i)
  int b = tid >> 12;
  int i = tid & (HWn - 1);
  const float* ps = src + (size_t)b * Cn * HWn + i;
  const float* pt = tgt + (size_t)b * Cn * HWn + i;
  float ss = 0.f, st = 0.f;
  #pragma unroll 4
  for (int c = 0; c < Cn; ++c) {
    float a = ps[(size_t)c * HWn]; ss += a * a;
    float d = pt[(size_t)c * HWn]; st += d * d;
  }
  float nsr = fmaxf(sqrtf(ss), 1e-12f);
  float ntg = fmaxf(sqrtf(st), 1e-12f);
  norm_src[tid] = nsr;
  inv_src[tid]  = 1.0f / nsr;
  inv_tgt[tid]  = 1.0f / ntg;
}

// ---------------------------------------------------------------------------
// Kernel N2: normalized + transposed f16 copies: [b][c][i] f32 -> [b][i][c] f16
// (32x32 LDS tile transpose; z dim selects src vs tgt and batch).
// ---------------------------------------------------------------------------
__global__ __launch_bounds__(256) void normalize_transpose_kernel(
    const float* __restrict__ src, const float* __restrict__ tgt,
    const float* __restrict__ inv_src, const float* __restrict__ inv_tgt,
    _Float16* __restrict__ srcT, _Float16* __restrict__ tgtT)
{
  __shared__ _Float16 tile[32][33];
  int z = blockIdx.z;
  int b = z & 7;
  bool is_src = (z < 8);
  const float* in  = is_src ? src : tgt;
  const float* inv = is_src ? inv_src : inv_tgt;
  _Float16*    out = is_src ? srcT : tgtT;

  int i0 = blockIdx.x * 32;
  int c0 = blockIdx.y * 32;
  int tx = threadIdx.x & 31;
  int ty = threadIdx.x >> 5;

  float invv = inv[b * HWn + i0 + tx];
  #pragma unroll
  for (int k = 0; k < 4; ++k) {
    int c = c0 + ty + 8 * k;
    float v = in[((size_t)b * Cn + c) * HWn + i0 + tx] * invv;
    tile[tx][ty + 8 * k] = (_Float16)v;
  }
  __syncthreads();
  #pragma unroll
  for (int k = 0; k < 4; ++k) {
    int il = ty + 8 * k;
    out[((size_t)b * HWn + i0 + il) * 256 + c0 + tx] = tile[il][tx];
  }
}

// ---------------------------------------------------------------------------
// Kernel G: fused correlation-GEMM + row-max + 1x1 conv + BN + SiLU.
// Workgroup = (batch, 128 rows of i).  8 waves, each owns 16 i rows.
// A fragments (16x32 f16, K=0..255) live in registers for the whole kernel.
// j-loop processes 2 tiles / iteration with independent accumulator chains
// so the WMMA->VALU hazard window of one chain hides under the other.
// ---------------------------------------------------------------------------
__global__ __launch_bounds__(256) void corr_conv_kernel(
    const _Float16* __restrict__ srcT, const _Float16* __restrict__ tgtT,
    const _Float16* __restrict__ Wh,   const float* __restrict__ norm_src,
    const float* __restrict__ w_scale, const float* __restrict__ w_wsum_s,
    const float* __restrict__ w_const, float* __restrict__ out)
{
  int b    = blockIdx.y;
  int i0   = blockIdx.x * 128;
  int t    = threadIdx.x;
  int wave = t >> 5;
  int lane = t & 31;
  int n    = lane & 15;   // A: row m; B: col n
  int half = lane >> 4;   // K-half selector
  int iw   = i0 + wave * 16;

  // ---- A fragments: rows iw..iw+15, K tiles 0..7 (k order used here, not the ISA table order:
  //      lane m=L&15; element s holds K = 16*half + s (matches B)
  //      => one contiguous 16-element (32B) run per lane per tile) ----
  FragH a[8];
  const _Float16* abase = srcT + ((size_t)b * HWn + iw + n) * 256;
  #pragma unroll
  for (int kt = 0; kt < 8; ++kt) {
    a[kt].q[0] = *(const uint4*)(abase + kt * 32 + 16 * half);
    a[kt].q[1] = *(const uint4*)(abase + kt * 32 + 16 * half + 8);
  }

  // ---- stream all j tiles (2 per iteration), fused running max over j ----
  float vmax[8];
  #pragma unroll
  for (int r = 0; r < 8; ++r) vmax[r] = -INFINITY;

  const _Float16* tbase = tgtT + ((size_t)b * HWn + n) * 256;
  for (int jt = 0; jt < HWn / 32; ++jt) {
    const _Float16* bb0 = tbase + (size_t)jt * 32 * 256;
    const _Float16* bb1 = bb0 + 16 * 256;
    v8f acc0 = {};
    v8f acc1 = {};
    #pragma unroll
    for (int kt = 0; kt < 8; ++kt) {
      // B uses the same contiguous k order as A (not the ISA table order): lane n=L&15, VGPR v holds K = 16*half + 2v
      // => one contiguous 16-element (32B) run per lane per tile
      FragH bf0, bf1;
      bf0.q[0] = *(const uint4*)(bb0 + kt * 32 + 16 * half);
      bf0.q[1] = *(const uint4*)(bb0 + kt * 32 + 16 * half + 8);
      bf1.q[0] = *(const uint4*)(bb1 + kt * 32 + 16 * half);
      bf1.q[1] = *(const uint4*)(bb1 + kt * 32 + 16 * half + 8);
      acc0 = __builtin_amdgcn_wmma_f32_16x16x32_f16(false, a[kt].h, false, bf0.h,
                                                    (short)0, acc0, false, false);
      acc1 = __builtin_amdgcn_wmma_f32_16x16x32_f16(false, a[kt].h, false, bf1.h,
                                                    (short)0, acc1, false, false);
    }
    #pragma unroll
    for (int r = 0; r < 8; ++r)
      vmax[r] = fmaxf(vmax[r], fmaxf(acc0[r], acc1[r]));
  }

  // Cross-lane max within each 16-lane group: C/D VGPR r holds M = r + 8*half,
  // shared by the 16 lanes of that half -> after reduction every lane holds the
  // row max for exactly the M rows its conv-output fragment will need.
  #pragma unroll
  for (int r = 0; r < 8; ++r) {
    #pragma unroll
    for (int off = 8; off >= 1; off >>= 1)
      vmax[r] = fmaxf(vmax[r], __shfl_xor(vmax[r], off, 16));
  }

  // per-row un-normalization factors (dot(W,src) = norm_i * dot(W,src_n))
  float normv[8];
  #pragma unroll
  for (int r = 0; r < 8; ++r)
    normv[r] = norm_src[(size_t)b * HWn + iw + 8 * half + r];

  // ---- fused 1x1 conv (reuses A fragments) + BN + SiLU ----
  for (int ot = 0; ot < 16; ++ot) {
    int o = ot * 16 + n;
    const _Float16* wb = Wh + (size_t)o * 256;
    v8f acc = {};
    #pragma unroll
    for (int kt = 0; kt < 8; ++kt) {
      FragH bf;
      bf.q[0] = *(const uint4*)(wb + kt * 32 + 16 * half);
      bf.q[1] = *(const uint4*)(wb + kt * 32 + 16 * half + 8);
      acc = __builtin_amdgcn_wmma_f32_16x16x32_f16(false, a[kt].h, false, bf.h,
                                                   (short)0, acc, false, false);
    }
    float sc = w_scale[o], ws = w_wsum_s[o], cs = w_const[o];
    float* op = out + ((size_t)b * OUTC + o) * HWn + iw + 8 * half;
    #pragma unroll
    for (int r = 0; r < 8; ++r) {
      float y = acc[r] * normv[r] * sc + vmax[r] * ws + cs;
      op[r] = y / (1.0f + __expf(-y));   // SiLU
    }
  }
}

// ---------------------------------------------------------------------------
extern "C" void kernel_launch(void* const* d_in, const int* in_sizes, int n_in,
                              void* d_out, int out_size, void* d_ws, size_t ws_size,
                              hipStream_t stream) {
  const float* src    = (const float*)d_in[0];
  const float* tgt    = (const float*)d_in[1];
  const float* conv_w = (const float*)d_in[2];
  const float* conv_b = (const float*)d_in[3];
  const float* gamma  = (const float*)d_in[4];
  const float* beta   = (const float*)d_in[5];
  const float* mean   = (const float*)d_in[6];
  const float* var    = (const float*)d_in[7];
  float* out = (float*)d_out;

  char* ws = (char*)d_ws;
  size_t off = 0;
  _Float16* srcT = (_Float16*)(ws + off); off += (size_t)Bn * HWn * 256 * 2; // 16 MB
  _Float16* tgtT = (_Float16*)(ws + off); off += (size_t)Bn * HWn * 256 * 2; // 16 MB
  float* norm_src = (float*)(ws + off);   off += (size_t)Bn * HWn * 4;
  float* inv_src  = (float*)(ws + off);   off += (size_t)Bn * HWn * 4;
  float* inv_tgt  = (float*)(ws + off);   off += (size_t)Bn * HWn * 4;
  _Float16* Wh    = (_Float16*)(ws + off); off += (size_t)OUTC * 256 * 2;
  float* w_scale  = (float*)(ws + off);   off += 1024;
  float* w_wsum_s = (float*)(ws + off);   off += 1024;
  float* w_const  = (float*)(ws + off);   off += 1024;

  prep_kernel<<<1, 256, 0, stream>>>(conv_w, conv_b, gamma, beta, mean, var,
                                     Wh, w_scale, w_wsum_s, w_const);
  norm_kernel<<<(Bn * HWn) / 256, 256, 0, stream>>>(src, tgt, norm_src,
                                                    inv_src, inv_tgt);
  normalize_transpose_kernel<<<dim3(HWn / 32, Cn / 32, 2 * Bn), 256, 0, stream>>>(
      src, tgt, inv_src, inv_tgt, srcT, tgtT);
  corr_conv_kernel<<<dim3(HWn / 128, Bn), 256, 0, stream>>>(
      srcT, tgtT, Wh, norm_src, w_scale, w_wsum_s, w_const, out);
}
